// MultiProductAttention_16114717295054
// MI455X (gfx1250) — matrix operand order repaired and checked by reading, not compiled, not run on hardware
//
#include <hip/hip_runtime.h>
#include <hip/hip_bf16.h>
#include <math.h>

// ---------------------------------------------------------------------------
// MultiProductAttention for MI455X (gfx1250, wave32, WMMA 16x16x32 bf16)
//   [1] convert x / weights to bf16 (kv|kvh|g concatenated to 320 rows)
//   [2] WMMA GEMM (TDM double-buffered): q = x @ Wq^T
//   [3] WMMA GEMM: kvcat = x @ [Wkv;Wkvh;Wg]^T
//   [4] sequential gating scan over N
//   [5] rope(q)*scale -> bf16 ; k/v outer products + rope(k) -> bf16
//   [6] flash attention (TDM K-tiles, online softmax, WMMA QK^T and PV)
//   [7] WMMA GEMM: out = o @ Wout^T -> f32
// ---------------------------------------------------------------------------

typedef __bf16 bf16_t;
typedef __attribute__((ext_vector_type(16))) __bf16 v16bf;
typedef __attribute__((ext_vector_type(8)))  __bf16 v8bf;
typedef __attribute__((ext_vector_type(8)))  float  v8f;
typedef __attribute__((ext_vector_type(4)))  unsigned int u32x4;
typedef __attribute__((ext_vector_type(4)))  unsigned int u32x4v;
typedef __attribute__((ext_vector_type(8)))  int i32x8v;
typedef __attribute__((ext_vector_type(4)))  int i32x4v;

#if __has_builtin(__builtin_amdgcn_tensor_load_to_lds) && \
    __has_builtin(__builtin_amdgcn_s_wait_tensorcnt)
#define HAVE_TDM 1
#else
#define HAVE_TDM 0
#endif

__device__ __forceinline__ v8f wmma_bf16(v16bf a, v16bf b, v8f c) {
  return __builtin_amdgcn_wmma_f32_16x16x32_bf16(false, a, false, b, (short)0, c,
                                                 false, false);
}

constexpr int Bb = 2, Nn = 2048, Ee = 2048, Hh = 16, Dd = 128;
constexpr int Mm = Bb * Nn;              // 4096 rows total
constexpr int NC = 320;                  // kv(256) | kvh(32) | g(32)
constexpr float QSCALE = 0.08838834764831845f;  // 1/sqrt(128)

#if HAVE_TDM
// Issue one TDM 2D tile load: tile_d1 rows of tile_d0 bf16 elems, global row
// stride `stride_elems` (elems), into LDS at byte offset lds_off. Optional LDS
// row padding: pad_interval/pad_amount are the encoded D# field values.
// D# packing per CDNA5 ISA (group0: count/lds/global/type;
// group1: data_size, pad, tensor/tile dims, dim0 stride).
__device__ __forceinline__ void tdm_load_2d(unsigned lds_off, const void* gaddr,
                                            unsigned tile_d0, unsigned tile_d1,
                                            unsigned stride_elems, bool pad_en,
                                            unsigned pad_interval,
                                            unsigned pad_amount) {
  unsigned long long ga = (unsigned long long)(uintptr_t)gaddr;
  u32x4v g0;
  g0[0] = 1u;                                          // count=1, user desc
  g0[1] = lds_off;                                     // lds_addr (bytes)
  g0[2] = (unsigned)ga;                                // global_addr[31:0]
  g0[3] = ((unsigned)(ga >> 32) & 0x01FFFFFFu) | (2u << 30);  // [56:32]|type=2
  i32x8v g1;
  unsigned w0 = (1u << 16);                            // data_size = 2 bytes
  if (pad_en)
    w0 |= (1u << 20) | (pad_interval << 22) | (pad_amount << 25);
  g1[0] = (int)w0;
  g1[1] = (int)((tile_d0 & 0xFFFFu) << 16);            // tensor_dim0 = tile_d0
  g1[2] = (int)(((tile_d0 >> 16) & 0xFFFFu) | ((tile_d1 & 0xFFFFu) << 16));
  g1[3] = (int)(((tile_d1 >> 16) & 0xFFFFu) | ((tile_d0 & 0xFFFFu) << 16));
  g1[4] = (int)(tile_d1 & 0xFFFFu);                    // tile_dim1; tile_dim2=0
  g1[5] = (int)stride_elems;                           // tensor_dim0_stride lo
  g1[6] = 0;
  g1[7] = 0;
  i32x4v z4;
  z4[0] = z4[1] = z4[2] = z4[3] = 0;
#if defined(__clang_major__) && __clang_major__ >= 23
  i32x8v z8;
#pragma unroll
  for (int i = 0; i < 8; ++i) z8[i] = 0;
  __builtin_amdgcn_tensor_load_to_lds(g0, g1, z4, z4, z8, 0);
#else
  __builtin_amdgcn_tensor_load_to_lds(g0, g1, z4, z4, 0);
#endif
}
__device__ __forceinline__ unsigned lds_off_of(const void* p) {
  return (unsigned)(unsigned long long)(uintptr_t)p;  // low 32 bits = LDS addr
}
#endif

// --------------------------- [1] convert ----------------------------------
__global__ void f32_to_bf16(const float* __restrict__ src,
                            bf16_t* __restrict__ dst, int n) {
  int i = blockIdx.x * blockDim.x + threadIdx.x;
  int stride = gridDim.x * blockDim.x;
  for (; i < n; i += stride) dst[i] = (bf16_t)src[i];
}

// --------------------------- [2,3,7] GEMM ----------------------------------
// C[M,N] = A[M,K] @ W[N,K]^T, bf16 in / f32 out. Block tile 128 x BN, 8 waves.
// A-frag: lanes 0-15 K=0-15, 16-31 K=16-31 (same k order as B); B-frag: lanes 0-15 K=0-15, 16-31 K=16-31.
// TDM path: double-buffered LDS, wave0 DMAs next K tile while all compute.
template <int BN, int WM, int WN>
__global__ __launch_bounds__(256) void gemm_bf16nt(const bf16_t* __restrict__ A,
                                                   const bf16_t* __restrict__ W,
                                                   float* __restrict__ C,
                                                   int M, int N, int K) {
  constexpr int BM = 128;
  constexpr int WAVES_N = BN / WN;
  constexpr int TM = WM / 16, TN = WN / 16;
  constexpr int AS = 40;  // LDS A row stride (elems): 64B row + 16B pad
  __shared__ alignas(32) bf16_t Asd[2 * BM * AS];
  __shared__ alignas(32) bf16_t Wsd[2 * BN * 32];

  const int tid = threadIdx.x;
  const int wid = tid >> 5, lane = tid & 31;
  const int wm = wid / WAVES_N, wn = wid % WAVES_N;
  const int half = lane >> 4, l16 = lane & 15;
  const size_t mbase = (size_t)blockIdx.y * BM;
  const size_t nbase = (size_t)blockIdx.x * BN;

  v8f acc[TM][TN];
#pragma unroll
  for (int i = 0; i < TM; ++i)
#pragma unroll
    for (int j = 0; j < TN; ++j)
#pragma unroll
      for (int e = 0; e < 8; ++e) acc[i][j][e] = 0.0f;

  auto compute_tile = [&](const bf16_t* as, const bf16_t* wsb) {
    v16bf af[TM], bfr[TN];
#pragma unroll
    for (int i = 0; i < TM; ++i) {
      int row = wm * WM + i * 16 + l16;
      v8bf lo = *(const v8bf*)&as[row * AS + 16 * half];
      v8bf hi = *(const v8bf*)&as[row * AS + 16 * half + 8];
      v16bf a;
#pragma unroll
      for (int e = 0; e < 8; ++e) { a[e] = lo[e]; a[e + 8] = hi[e]; }
      af[i] = a;
    }
#pragma unroll
    for (int j = 0; j < TN; ++j) {
      int row = wn * WN + j * 16 + l16;
      bfr[j] = *(const v16bf*)&wsb[row * 32 + 16 * half];
    }
#pragma unroll
    for (int i = 0; i < TM; ++i)
#pragma unroll
      for (int j = 0; j < TN; ++j)
        acc[i][j] = wmma_bf16(af[i], bfr[j], acc[i][j]);
  };

#if HAVE_TDM
  const unsigned as_off = lds_off_of(Asd);
  const unsigned ws_off = lds_off_of(Wsd);
  auto issue_tiles = [&](int k0, int buf) {
    // A tile: 128 rows x 32 elems, pad each 64B row by 16B -> stride 80B.
    // pad_interval code 3 = 16 DWORDs (64B); pad_amount code 3 = 4 DWORDs.
    tdm_load_2d(as_off + (unsigned)buf * (BM * AS * 2),
                A + mbase * (size_t)K + k0, 32, BM, (unsigned)K, true, 3, 3);
    // W tile: BN rows x 32 elems, unpadded (stride 64B).
    tdm_load_2d(ws_off + (unsigned)buf * (BN * 32 * 2),
                W + nbase * (size_t)K + k0, 32, BN, (unsigned)K, false, 0, 0);
  };
  const int KT = K / 32;
  if (wid == 0) issue_tiles(0, 0);
  for (int kt = 0; kt < KT; ++kt) {
    const int cur = kt & 1;
    if (wid == 0) __builtin_amdgcn_s_wait_tensorcnt(0);
    __syncthreads();  // tile `cur` ready; prior reads of `cur^1` drained
    if (wid == 0 && kt + 1 < KT) issue_tiles((kt + 1) * 32, cur ^ 1);
    compute_tile(&Asd[cur * BM * AS], &Wsd[cur * BN * 32]);
  }
#else
  for (int k0 = 0; k0 < K; k0 += 32) {
#pragma unroll
    for (int p = 0; p < BM / 64; ++p) {
      int r = p * 64 + (tid >> 2);
      int c8 = (tid & 3) * 8;
      *(u32x4*)&Asd[r * AS + c8] =
          *(const u32x4*)&A[(mbase + r) * (size_t)K + k0 + c8];
    }
#pragma unroll
    for (int p = 0; p < BN / 64; ++p) {
      int r = p * 64 + (tid >> 2);
      int c8 = (tid & 3) * 8;
      *(u32x4*)&Wsd[r * 32 + c8] =
          *(const u32x4*)&W[(nbase + r) * (size_t)K + k0 + c8];
    }
    __syncthreads();
    compute_tile(Asd, Wsd);
    __syncthreads();
  }
#endif

  // C-layout: VGPR r -> row r + 8*half, col = lane&15
#pragma unroll
  for (int i = 0; i < TM; ++i)
#pragma unroll
    for (int j = 0; j < TN; ++j)
#pragma unroll
      for (int r = 0; r < 8; ++r) {
        size_t rg = mbase + wm * WM + i * 16 + r + 8 * half;
        size_t cg = nbase + wn * WN + j * 16 + l16;
        C[rg * (size_t)N + cg] = acc[i][j][r];
      }
}

// --------------------------- [4] gating scan -------------------------------
// kvcat columns: [0,128) kp | [128,256) vp | [256,272) kh | [272,288) vh |
// [288,320) g.
__global__ void gate_scan(float* __restrict__ kvcat) {
  int t = blockIdx.x * blockDim.x + threadIdx.x;
  if (t >= Bb * 2 * Hh) return;
  int b = t >> 5, ch = t & 31;
  float lfc = 0.0f, s = 0.0f;
  for (int n = 0; n < Nn; ++n) {
    size_t m = (size_t)(b * Nn + n) * NC;
    float gv = kvcat[m + 288 + ch];
    float kvh = kvcat[m + 256 + ch];
    s += kvh * __expf(lfc);            // decayed cumsum
    kvcat[m + 256 + ch] = s;
    float ls = (gv >= 0.0f) ? -log1pf(__expf(-gv))
                            : (gv - log1pf(__expf(gv)));
    lfc += ls;                         // applied from the next position
  }
}

// --------------------------- [5] rope / kv build ---------------------------
__global__ void rope_q_to_bf16(const float* __restrict__ q,
                               bf16_t* __restrict__ qb) {
  int idx = blockIdx.x * blockDim.x + threadIdx.x;  // over Mm*Hh*64
  if (idx >= Mm * Hh * 64) return;
  int j = idx & 63;
  int h = (idx >> 6) & 15;
  int m = idx >> 10;
  int n = m & (Nn - 1);
  size_t base = ((size_t)m * Hh + h) * Dd;
  float inv = __powf(10000.0f, -(float)j * (1.0f / 64.0f));
  float ang = (float)n * inv;
  float sn, cs;
  __sincosf(ang, &sn, &cs);
  float t1 = q[base + j], t2 = q[base + j + 64];
  qb[base + j]      = (bf16_t)((t1 * cs - t2 * sn) * QSCALE);
  qb[base + j + 64] = (bf16_t)((t2 * cs + t1 * sn) * QSCALE);
}

__global__ void build_kv(const float* __restrict__ kvcat,
                         bf16_t* __restrict__ kb, bf16_t* __restrict__ vb) {
  int idx = blockIdx.x * blockDim.x + threadIdx.x;
  if (idx >= Mm * Hh * 64) return;
  int j = idx & 63;
  int h = (idx >> 6) & 15;
  int m = idx >> 10;
  int n = m & (Nn - 1);
  const float* row = kvcat + (size_t)m * NC;
  float kh = row[256 + h], vh = row[272 + h];
  float k1 = row[j] * kh, k2 = row[j + 64] * kh;
  float inv = __powf(10000.0f, -(float)j * (1.0f / 64.0f));
  float ang = (float)n * inv;
  float sn, cs;
  __sincosf(ang, &sn, &cs);
  size_t base = ((size_t)m * Hh + h) * Dd;
  kb[base + j]      = (bf16_t)(k1 * cs - k2 * sn);
  kb[base + j + 64] = (bf16_t)(k2 * cs + k1 * sn);
  vb[base + j]      = (bf16_t)(row[128 + j] * vh);
  vb[base + j + 64] = (bf16_t)(row[128 + j + 64] * vh);
}

// --------------------------- [6] flash attention ---------------------------
// grid (N/128, H, B); 256 threads = 8 waves, each wave owns 16 q rows.
// K tile DMA'd by TDM (padded rows); V transposed into LDS for PV B-frags.
__global__ __launch_bounds__(256) void flash_attn(const bf16_t* __restrict__ qb,
                                                  const bf16_t* __restrict__ kb,
                                                  const bf16_t* __restrict__ vb,
                                                  bf16_t* __restrict__ ob) {
  constexpr int KS = 144;  // Ks row stride (elems) -> 288B, 32B aligned
  __shared__ alignas(32) bf16_t Ks[32 * KS];
  __shared__ alignas(32) bf16_t Vt[128 * 32];
  __shared__ alignas(32) bf16_t Ps[8 * 16 * 40];

  const int tid = threadIdx.x, wid = tid >> 5, lane = tid & 31;
  const int half = lane >> 4, l16 = lane & 15;
  const int b = blockIdx.z, h = blockIdx.y;
  const int qbase = blockIdx.x * 128;
  const int q0 = qbase + wid * 16;

  // q fragments (4 chunks of K=32 over D=128), loaded straight from global
  v16bf qf[4];
  {
    size_t base = ((size_t)(b * Nn + q0 + l16) * Hh + h) * Dd;
#pragma unroll
    for (int c = 0; c < 4; ++c) {
      v8bf lo = *(const v8bf*)&qb[base + c * 32 + 16 * half];
      v8bf hi = *(const v8bf*)&qb[base + c * 32 + 16 * half + 8];
      v16bf a;
#pragma unroll
      for (int e = 0; e < 8; ++e) { a[e] = lo[e]; a[e + 8] = hi[e]; }
      qf[c] = a;
    }
  }

  v8f o[8];
#pragma unroll
  for (int dt = 0; dt < 8; ++dt)
#pragma unroll
    for (int e = 0; e < 8; ++e) o[dt][e] = 0.0f;
  float mrow[8], lrow[8];
#pragma unroll
  for (int r = 0; r < 8; ++r) { mrow[r] = -3.0e38f; lrow[r] = 0.0f; }

  const int ntiles = (qbase + 128) / 32;
  for (int t = 0; t < ntiles; ++t) {
    const int j0 = t * 32;
#if HAVE_TDM
    // K tile 32x128 via TDM; pad each 256B row by 32B -> LDS stride 288B.
    // pad_interval code 5 = 64 DWORDs (256B); pad_amount code 7 = 8 DWORDs.
    if (wid == 0)
      tdm_load_2d(lds_off_of(Ks), &kb[((size_t)(b * Nn + j0) * Hh + h) * Dd],
                  128, 32, Hh * Dd, true, 5, 7);
#endif
    // V tile load + transpose into Vt[d][key] (all threads)
#pragma unroll
    for (int e = 0; e < 2; ++e) {
      int lin = tid + e * 256;       // 0..511
      int row = lin >> 4;            // key 0..31
      int c8 = (lin & 15) * 8;       // d 0..120
      size_t g = ((size_t)(b * Nn + j0 + row) * Hh + h) * Dd + c8;
#if !HAVE_TDM
      *(u32x4*)&Ks[row * KS + c8] = *(const u32x4*)&kb[g];
#endif
      v8bf vv = *(const v8bf*)&vb[g];
#pragma unroll
      for (int z = 0; z < 8; ++z) Vt[(c8 + z) * 32 + row] = vv[z];
    }
#if HAVE_TDM
    if (wid == 0) __builtin_amdgcn_s_wait_tensorcnt(0);
#endif
    __syncthreads();

    if (j0 <= q0 + 15) {  // wave-uniform: some key in tile is unmasked
      // S = q @ K^T : two 16x16 tiles over 32 keys
      v8f sacc[2];
#pragma unroll
      for (int jt = 0; jt < 2; ++jt) {
        v8f a;
#pragma unroll
        for (int e = 0; e < 8; ++e) a[e] = 0.0f;
#pragma unroll
        for (int c = 0; c < 4; ++c) {
          v16bf bk =
              *(const v16bf*)&Ks[(jt * 16 + l16) * KS + c * 32 + 16 * half];
          a = wmma_bf16(qf[c], bk, a);
        }
        sacc[jt] = a;
      }

      // online softmax over rows r+8*half; 16-lane xor-shuffle reductions
      float resc[8];
      bf16_t* pw = &Ps[wid * 16 * 40];
#pragma unroll
      for (int r = 0; r < 8; ++r) {
        int qr = q0 + r + 8 * half;
        float s0 = ((j0 + l16) <= qr) ? sacc[0][r] : -3.0e38f;
        float s1 = ((j0 + 16 + l16) <= qr) ? sacc[1][r] : -3.0e38f;
        float mx = fmaxf(s0, s1);
#pragma unroll
        for (int msk = 1; msk < 16; msk <<= 1)
          mx = fmaxf(mx, __shfl_xor(mx, msk, 32));
        float mnew = fmaxf(mrow[r], mx);
        float p0 = __expf(s0 - mnew), p1 = __expf(s1 - mnew);
        float rs = p0 + p1;
#pragma unroll
        for (int msk = 1; msk < 16; msk <<= 1) rs += __shfl_xor(rs, msk, 32);
        float rc = __expf(mrow[r] - mnew);
        lrow[r] = lrow[r] * rc + rs;
        mrow[r] = mnew;
        resc[r] = rc;
        int prow = r + 8 * half;
        pw[prow * 40 + l16] = (bf16_t)p0;        // keys j0..j0+15
        pw[prow * 40 + 16 + l16] = (bf16_t)p1;   // keys j0+16..j0+31
      }
      // per-wave LDS RAW: DS ops in-order; wait before fragment reads
      asm volatile("s_wait_dscnt 0" ::: "memory");

      // P as A-fragment (16 q rows x 32 keys)
      v16bf pf;
      {
        v8bf lo = *(const v8bf*)&pw[l16 * 40 + 16 * half];
        v8bf hi = *(const v8bf*)&pw[l16 * 40 + 16 * half + 8];
#pragma unroll
        for (int e = 0; e < 8; ++e) { pf[e] = lo[e]; pf[e + 8] = hi[e]; }
      }

      // o = o*resc + P @ V  (8 WMMAs across D=128)
#pragma unroll
      for (int dt = 0; dt < 8; ++dt) {
        v8f oc = o[dt];
#pragma unroll
        for (int r = 0; r < 8; ++r) oc[r] *= resc[r];
        v16bf bv = *(const v16bf*)&Vt[(dt * 16 + l16) * 32 + 16 * half];
        o[dt] = wmma_bf16(pf, bv, oc);
      }
    }
    __syncthreads();
  }

  // epilogue: normalize and write o as bf16 (layout m*2048 + h*128 + d)
#pragma unroll
  for (int dt = 0; dt < 8; ++dt)
#pragma unroll
    for (int r = 0; r < 8; ++r) {
      size_t row = (size_t)(b * Nn) + q0 + r + 8 * half;
      ob[(row * Hh + h) * Dd + dt * 16 + l16] =
          (bf16_t)(o[dt][r] / lrow[r]);
    }
}

// --------------------------- launch ----------------------------------------
extern "C" void kernel_launch(void* const* d_in, const int* in_sizes, int n_in,
                              void* d_out, int out_size, void* d_ws,
                              size_t ws_size, hipStream_t stream) {
  (void)in_sizes; (void)n_in; (void)out_size; (void)ws_size;
  const float* x    = (const float*)d_in[0];
  const float* Wq   = (const float*)d_in[1];
  const float* Wkv  = (const float*)d_in[2];
  const float* Wkvh = (const float*)d_in[3];
  const float* Wg   = (const float*)d_in[4];
  const float* Wout = (const float*)d_in[5];
  float* out = (float*)d_out;

  char* ws = (char*)d_ws;
  size_t off = 0;
  auto carve = [&](size_t bytes) -> char* {
    char* p = ws + off;
    off = (off + bytes + 255) & ~(size_t)255;
    return p;
  };

  bf16_t* xb    = (bf16_t*)carve((size_t)Mm * Ee * 2);
  bf16_t* Wqb   = (bf16_t*)carve((size_t)Hh * Dd * Ee * 2);
  bf16_t* Wcb   = (bf16_t*)carve((size_t)NC * Ee * 2);   // kv|kvh|g
  bf16_t* Woutb = (bf16_t*)carve((size_t)Ee * Hh * Dd * 2);
  float*  qf32  = (float*)carve((size_t)Mm * Hh * Dd * 4);
  float*  kvcat = (float*)carve((size_t)Mm * NC * 4);
  bf16_t* qbuf  = (bf16_t*)carve((size_t)Mm * Hh * Dd * 2);
  bf16_t* kbuf  = (bf16_t*)carve((size_t)Mm * Hh * Dd * 2);
  bf16_t* vbuf  = (bf16_t*)carve((size_t)Mm * Hh * Dd * 2);
  bf16_t* obuf  = (bf16_t*)carve((size_t)Mm * Hh * Dd * 2);

  // [1] converts
  f32_to_bf16<<<2048, 256, 0, stream>>>(x, xb, Mm * Ee);
  f32_to_bf16<<<2048, 256, 0, stream>>>(Wq, Wqb, Hh * Dd * Ee);
  f32_to_bf16<<<512, 256, 0, stream>>>(Wkv, Wcb, 2 * Dd * Ee);
  f32_to_bf16<<<256, 256, 0, stream>>>(Wkvh, Wcb + (size_t)256 * Ee, 2 * Hh * Ee);
  f32_to_bf16<<<256, 256, 0, stream>>>(Wg, Wcb + (size_t)288 * Ee, 2 * Hh * Ee);
  f32_to_bf16<<<2048, 256, 0, stream>>>(Wout, Woutb, Ee * Hh * Dd);

  // [2] q projection
  gemm_bf16nt<128, 64, 32>
      <<<dim3((Hh * Dd) / 128, Mm / 128), 256, 0, stream>>>(
          xb, Wqb, qf32, Mm, Hh * Dd, Ee);
  // [3] kv|kvh|g projection (N=320, 128x64 block tiles)
  gemm_bf16nt<64, 32, 32>
      <<<dim3(NC / 64, Mm / 128), 256, 0, stream>>>(
          xb, Wcb, kvcat, Mm, NC, Ee);

  // [4] gating scan
  gate_scan<<<1, 64, 0, stream>>>(kvcat);

  // [5] rope + kv build
  rope_q_to_bf16<<<(Mm * Hh * 64) / 256, 256, 0, stream>>>(qf32, qbuf);
  build_kv<<<(Mm * Hh * 64) / 256, 256, 0, stream>>>(kvcat, kbuf, vbuf);

  // [6] flash attention
  flash_attn<<<dim3(Nn / 128, Hh, Bb), 256, 0, stream>>>(qbuf, kbuf, vbuf, obuf);

  // [7] output projection -> f32 d_out
  gemm_bf16nt<128, 64, 32>
      <<<dim3(Ee / 128, Mm / 128), 256, 0, stream>>>(
          obuf, Woutb, out, Mm, Ee, Hh * Dd);
}
